// Model_58617713656142
// MI455X (gfx1250) — matrix operand order repaired and checked by reading, not compiled, not run on hardware
//
#include <hip/hip_runtime.h>
#include <hip/hip_bf16.h>

typedef __attribute__((ext_vector_type(16))) _Float16 v16h;
typedef __attribute__((ext_vector_type(8)))  _Float16 v8h;
typedef __attribute__((ext_vector_type(8)))  float    v8f;

#define CONV_THREADS 128   // 4 waves (wave32)
#define CONV_BLKN    256   // pixels (GEMM-N) per block
#define MTILE        32    // Cout rows per block

// ---------------------------------------------------------------------------
// Implicit-GEMM 3x3 conv (pad=1) via v_wmma_f32_16x16x32_f16.
//   in   : f16, spatially padded [Nb, Cinp, H+2, W+2], border zeroed,
//          Cinp % 32 == 0 (pad channels killed by zero-packed weights).
//   wp   : f16 packed [Mpad][9*Cinp], K = (ky*3+kx)*Cinp + ci, zero-filled.
//   out16: f16 padded [Nb, Cstore, Hout+2, Wout+2], or out32: f32 NCHW.
// Block tile 32(M) x 256(N); wave = 2 M-subtiles x 4 N-subtiles = 8 WMMA/chunk.
// Double-buffered LDS + split load/store staging: one barrier per K-chunk,
// global loads for chunk k+1 issued before the WMMAs of chunk k. All fragment
// ds_loads are issued as one clause so the 8 WMMAs run back-to-back.
// ---------------------------------------------------------------------------
template <int HIN, int WIN, int STRIDE>
__global__ __launch_bounds__(CONV_THREADS)
void conv3x3_wmma(const _Float16* __restrict__ in, const _Float16* __restrict__ wp,
                  const float* __restrict__ bias,
                  _Float16* __restrict__ out16, float* __restrict__ out32,
                  int Cinp, int Cout, int Cstore, int relu)
{
    constexpr int HOUT = (STRIDE == 2) ? HIN / 2 : HIN;
    constexpr int WOUT = (STRIDE == 2) ? WIN / 2 : WIN;
    constexpr int HP = HIN + 2, WP = WIN + 2, HPWP = HP * WP;
    constexpr int HW = HOUT * WOUT;
    constexpr int HOP = HOUT + 2, WOP = WOUT + 2;

    __shared__ alignas(64) _Float16 As[2][MTILE][32];
    __shared__ alignas(64) _Float16 Bs[2][CONV_BLKN][32];

    const int tid   = threadIdx.x;
    const int lane  = tid & 31;
    const int wave  = tid >> 5;
    const int mBase = blockIdx.y * MTILE;
    const int nBase = blockIdx.x * CONV_BLKN;
    const int K9    = 9 * Cinp;
    const int Cc    = Cinp >> 5;
    const int ksteps = 9 * Cc;

    // Staging pixel decode (grid sized so p < Ngemm always; HW/WOUT constexpr).
    int nimg[2], iy0[2], ix0[2];
#pragma unroll
    for (int s = 0; s < 2; ++s) {
        int p  = nBase + tid + s * CONV_THREADS;
        int n  = p / HW;
        int r  = p - n * HW;
        int oy = r / WOUT;
        int ox = r - oy * WOUT;
        nimg[s] = n;
        iy0[s]  = oy * STRIDE;        // padded row base: + ky in [0, HP-1]
        ix0[s]  = ox * STRIDE;
    }

    // A-stage mapping: each thread moves 8 CONTIGUOUS halves (b128 in/out).
    const int arow = tid >> 2;             // 0..31
    const int acol = (tid & 3) << 3;       // 0,8,16,24
    const _Float16* wrow = wp + (mBase + arow) * K9 + acol;

    // Pipeline registers for the next chunk.
    v8h awn;
    v8h btn[2][4];

    auto loadNext = [&](int r_, int cc_) {
        const int ky  = r_ / 3;
        const int kx  = r_ - ky * 3;
        const int ci0 = cc_ << 5;
        awn = *(const v8h*)&wrow[r_ * Cinp + ci0];
#pragma unroll
        for (int s = 0; s < 2; ++s) {
            const _Float16* bp = in +
                ((nimg[s] * Cinp + ci0) * HP + iy0[s] + ky) * WP + ix0[s] + kx;
#pragma unroll
            for (int v = 0; v < 4; ++v)
#pragma unroll
                for (int e = 0; e < 8; ++e)
                    btn[s][v][e] = bp[(v * 8 + e) * HPWP];
        }
    };
    auto storeNext = [&](int pb) {
        *(v8h*)&As[pb][arow][acol] = awn;
#pragma unroll
        for (int s = 0; s < 2; ++s)
#pragma unroll
            for (int v = 0; v < 4; ++v)
                *(v8h*)&Bs[pb][tid + s * CONV_THREADS][v * 8] = btn[s][v];
    };

    v8f acc[8] = {};
    const int mrow = lane & 15;
    const int hi   = lane >> 4;
    const int kb   = hi << 4;

    auto compute = [&](int pb) {
        // Fragments: A and B both use contiguous k = 16*hi + s (not the ISA table order).
        // Preload everything so the ds_loads clause and WMMAs run back-to-back.
        union Frag { v16h v; v8h h[2]; } af[2];
        v16h bf[4];
#pragma unroll
        for (int mi = 0; mi < 2; ++mi) {
            af[mi].h[0] = *(const v8h*)&As[pb][mi * 16 + mrow][hi * 16];
            af[mi].h[1] = *(const v8h*)&As[pb][mi * 16 + mrow][hi * 16 + 8];
        }
#pragma unroll
        for (int sub = 0; sub < 4; ++sub)
            bf[sub] = *(const v16h*)&Bs[pb][wave * 64 + sub * 16 + mrow][kb];
#pragma unroll
        for (int sub = 0; sub < 4; ++sub)
#pragma unroll
            for (int mi = 0; mi < 2; ++mi)
                acc[mi * 4 + sub] = __builtin_amdgcn_wmma_f32_16x16x32_f16(
                    false, af[mi].v, false, bf[sub], (short)0, acc[mi * 4 + sub], false, false);
    };

    // Prologue: stage chunk 0 into buffer 0.
    loadNext(0, 0);
    storeNext(0);
    int r = 0, cc = 0;
    for (int kk = 0; kk < ksteps; ++kk) {
        __syncthreads();
        int rn = r, cn = cc + 1;
        if (cn == Cc) { cn = 0; rn = r + 1; }
        const bool more = (kk + 1 < ksteps);
        if (more) loadNext(rn, cn);       // global loads overlap the WMMAs
        compute(kk & 1);
        if (more) storeNext((kk + 1) & 1);
        r = rn; cc = cn;
    }

    // ---- epilogue: D layout: VGPR i -> M = i + 8*(lane>=16), N = lane&15 ----
    const int ncol0 = lane & 15;
    const int moff  = (lane >> 4) * 8;
#pragma unroll
    for (int sub = 0; sub < 4; ++sub) {
        int p  = nBase + wave * 64 + sub * 16 + ncol0;
        int n  = p / HW;
        int rr = p - n * HW;
        int oy = rr / WOUT;
        int ox = rr - oy * WOUT;
#pragma unroll
        for (int mi = 0; mi < 2; ++mi) {
#pragma unroll
            for (int i = 0; i < 8; ++i) {
                int co = mBase + mi * 16 + moff + i;
                if (co < Cout) {
                    float v = acc[mi * 4 + sub][i] + bias[co];
                    if (relu) v = fmaxf(v, 0.0f);
                    if (out32)
                        out32[((n * Cout + co) * HOUT + oy) * WOUT + ox] = v;
                    else
                        out16[((n * Cstore + co) * HOP + oy + 1) * WOP + ox + 1] = (_Float16)v;
                }
            }
        }
    }
}

// --------------------------- helper kernels --------------------------------

// Zero the 1-pixel spatial border of a padded f16 buffer [N,C,H+2,W+2].
__global__ void k_zero_border(_Float16* __restrict__ buf, int N, int C, int H, int W)
{
    int Hp = H + 2, Wp = W + 2;
    int P = 2 * Wp + 2 * H;
    int total = N * C * P;
    int idx = blockIdx.x * blockDim.x + threadIdx.x;
    if (idx >= total) return;
    int nc = idx / P;
    int b  = idx - nc * P;
    int y, x;
    if (b < Wp)           { y = 0;      x = b; }
    else if (b < 2 * Wp)  { y = Hp - 1; x = b - Wp; }
    else { int t = b - 2 * Wp; y = 1 + (t >> 1); x = (t & 1) ? (Wp - 1) : 0; }
    buf[(nc * Hp + y) * Wp + x] = (_Float16)0.0f;
}

// f32 unpadded [N,C,H,W] -> f16 padded dst at channel offset coff (Cd alloc ch).
__global__ void k_cvt_pad(const float* __restrict__ src, _Float16* __restrict__ dst,
                          int N, int C, int H, int W, int coff, int Cd)
{
    int total = N * C * H * W;
    int idx = blockIdx.x * blockDim.x + threadIdx.x;
    if (idx >= total) return;
    int n = idx / (C * H * W);
    int r = idx - n * C * H * W;
    int c = r / (H * W);
    int q = r - c * (H * W);
    int y = q / W, x = q - y * W;
    dst[((n * Cd + coff + c) * (H + 2) + y + 1) * (W + 2) + x + 1] = (_Float16)src[idx];
}

// f16 padded -> f16 padded channel copy (interiors only).
__global__ void k_copy16(const _Float16* __restrict__ src, _Float16* __restrict__ dst,
                         int N, int C, int H, int W, int Cs, int coff, int Cd)
{
    int total = N * C * H * W;
    int idx = blockIdx.x * blockDim.x + threadIdx.x;
    if (idx >= total) return;
    int n = idx / (C * H * W);
    int r = idx - n * C * H * W;
    int c = r / (H * W);
    int q = r - c * (H * W);
    int y = q / W, x = q - y * W;
    int Hp = H + 2, Wp = W + 2;
    dst[((n * Cd + coff + c) * Hp + y + 1) * Wp + x + 1] =
        src[((n * Cs + c) * Hp + y + 1) * Wp + x + 1];
}

// backward_warp (border-clamped bilinear), reference algebra:
// xs = clamp(x + fx*W/(W-1), 0, W-1); ys likewise. f32 in, f16 padded out.
__global__ void k_warp16(const float* __restrict__ frame, const float* __restrict__ flow,
                         _Float16* __restrict__ dst, int N, int C, int H, int W,
                         int coff, int Cd)
{
    int HW = H * W;
    int idx = blockIdx.x * blockDim.x + threadIdx.x;
    if (idx >= N * HW) return;
    int n = idx / HW;
    int r = idx - n * HW;
    int y = r / W, x = r - y * W;
    float fx = flow[(n * 2 + 0) * HW + r];
    float fy = flow[(n * 2 + 1) * HW + r];
    float xs = fminf(fmaxf((float)x + fx * (float)W / (float)(W - 1), 0.0f), (float)(W - 1));
    float ys = fminf(fmaxf((float)y + fy * (float)H / (float)(H - 1), 0.0f), (float)(H - 1));
    float x0f = floorf(xs), y0f = floorf(ys);
    int x0 = (int)x0f, y0 = (int)y0f;
    int x1 = min(x0 + 1, W - 1), y1 = min(y0 + 1, H - 1);
    float wx = xs - x0f, wy = ys - y0f;
    float w00 = (1.0f - wx) * (1.0f - wy), w01 = wx * (1.0f - wy);
    float w10 = (1.0f - wx) * wy,          w11 = wx * wy;
    int Hp = H + 2, Wp = W + 2;
    for (int c = 0; c < C; ++c) {
        const float* fp = frame + (n * C + c) * HW;
        float v = fp[y0 * W + x0] * w00 + fp[y0 * W + x1] * w01 +
                  fp[y1 * W + x0] * w10 + fp[y1 * W + x1] * w11;
        dst[((n * Cd + coff + c) * Hp + y + 1) * Wp + x + 1] = (_Float16)v;
    }
}

// 2x bilinear upsample (half-pixel), f16 padded [N,C,H+2,W+2] -> [N,C,2H+2,2W+2].
__global__ void k_up2x16(const _Float16* __restrict__ src, _Float16* __restrict__ dst,
                         int N, int C, int H, int W)
{
    int Ho = 2 * H, Wo = 2 * W;
    int total = N * C * Ho * Wo;
    int idx = blockIdx.x * blockDim.x + threadIdx.x;
    if (idx >= total) return;
    int nc = idx / (Ho * Wo);
    int r  = idx - nc * (Ho * Wo);
    int yo = r / Wo, xo = r - yo * Wo;
    float xs = fminf(fmaxf(((float)xo + 0.5f) * 0.5f - 0.5f, 0.0f), (float)(W - 1));
    float ys = fminf(fmaxf(((float)yo + 0.5f) * 0.5f - 0.5f, 0.0f), (float)(H - 1));
    int x0 = (int)xs, y0 = (int)ys;
    int x1 = min(x0 + 1, W - 1), y1 = min(y0 + 1, H - 1);
    float wx = xs - (float)x0, wy = ys - (float)y0;
    int Hp = H + 2, Wp = W + 2;
    const _Float16* sp = src + nc * Hp * Wp;
    float v = (float)sp[(y0 + 1) * Wp + x0 + 1] * (1.0f - wx) * (1.0f - wy) +
              (float)sp[(y0 + 1) * Wp + x1 + 1] * wx * (1.0f - wy) +
              (float)sp[(y1 + 1) * Wp + x0 + 1] * (1.0f - wx) * wy +
              (float)sp[(y1 + 1) * Wp + x1 + 1] * wx * wy;
    dst[(nc * (Ho + 2) + yo + 1) * (Wo + 2) + xo + 1] = (_Float16)v;
}

// Pack OIHW f32 weights -> f16 [Mpad][9*Cinp], K=(ky*3+kx)*Cinp+ci, zero-fill.
__global__ void k_pack_w(const float* __restrict__ w, _Float16* __restrict__ dst,
                         int Cout, int Cin, int Cinp, int Mpad)
{
    int K9 = 9 * Cinp;
    int total = Mpad * K9;
    int idx = blockIdx.x * blockDim.x + threadIdx.x;
    if (idx >= total) return;
    int co = idx / K9;
    int rr = idx - co * K9;
    int r  = rr / Cinp;
    int ci = rr - r * Cinp;
    _Float16 v = (_Float16)0.0f;
    if (co < Cout && ci < Cin) v = (_Float16)w[(co * Cin + ci) * 9 + r];
    dst[idx] = v;
}

// ---------------------------------------------------------------------------
extern "C" void kernel_launch(void* const* d_in, const int* in_sizes, int n_in,
                              void* d_out, int out_size, void* d_ws, size_t ws_size,
                              hipStream_t stream)
{
    (void)in_sizes; (void)n_in; (void)out_size; (void)ws_size;
    const int N = 4;
    const int H1 = 384, W1 = 384, H2 = 192, W2 = 192, H4 = 96, W4 = 96, H8 = 48, W8 = 48;
    const size_t P1 = 386 * 386, P2 = 194 * 194, P4 = 98 * 98, P8 = 50 * 50;

    const float* img0   = (const float*)d_in[0];
    const float* img1   = (const float*)d_in[1];
    const float* ft0    = (const float*)d_in[2];
    const float* ft1    = (const float*)d_in[3];
    const float* ft0_s2 = (const float*)d_in[4];
    const float* ft1_s2 = (const float*)d_in[5];
    const float* ft0_s4 = (const float*)d_in[6];
    const float* ft1_s4 = (const float*)d_in[7];
    const float* c0_0   = (const float*)d_in[8];
    const float* c0_1   = (const float*)d_in[9];
    const float* c0_2   = (const float*)d_in[10];
    const float* c1_0   = (const float*)d_in[11];
    const float* c1_1   = (const float*)d_in[12];
    const float* c1_2   = (const float*)d_in[13];
    const float* wd0a = (const float*)d_in[14]; const float* bd0a = (const float*)d_in[15];
    const float* wd0b = (const float*)d_in[16]; const float* bd0b = (const float*)d_in[17];
    const float* wd1a = (const float*)d_in[18]; const float* bd1a = (const float*)d_in[19];
    const float* wd1b = (const float*)d_in[20]; const float* bd1b = (const float*)d_in[21];
    const float* wd2a = (const float*)d_in[22]; const float* bd2a = (const float*)d_in[23];
    const float* wd2b = (const float*)d_in[24]; const float* bd2b = (const float*)d_in[25];
    const float* wu0  = (const float*)d_in[26]; const float* bu0  = (const float*)d_in[27];
    const float* wu1  = (const float*)d_in[28]; const float* bu1  = (const float*)d_in[29];
    const float* wu2  = (const float*)d_in[30]; const float* bu2  = (const float*)d_in[31];
    const float* wfin = (const float*)d_in[32]; const float* bfin = (const float*)d_in[33];

    // ---- workspace arenas (f16 padded activations) ----
    char* wsb = (char*)d_ws;
    size_t off = 0;
    auto alloc = [&](size_t halves) -> _Float16* {
        off = (off + 255) & ~(size_t)255;
        _Float16* p = (_Float16*)(wsb + off);
        off += halves * sizeof(_Float16);
        return p;
    };
    _Float16* bufA = alloc((size_t)96  * N * P1); // s0_in(96) / up2(64)
    _Float16* bufB = alloc((size_t)96  * N * P1); // s1_in(128@P2) / cat2(64@P2) / x2(96@P1)
    _Float16* s0b  = alloc((size_t)32  * N * P2);
    _Float16* bufD = alloc((size_t)128 * N * P2); // t0(32@P2) / s2_in(192@P4) / up1(128@P2)
    _Float16* bufE = alloc((size_t)128 * N * P4); // t1(64) / u0_in(96) / cat1(128)
    _Float16* bufF = alloc((size_t)32  * N * P2); // t2(96@P8) / x0(64@P4) / x1(32@P2)
    _Float16* s1b  = alloc((size_t)64  * N * P4);
    _Float16* s2b  = alloc((size_t)96  * N * P8);

    _Float16* wpA = alloc(600000);
    _Float16* pwd0a = wpA;            // 32 x 9*96
    _Float16* pwd0b = pwd0a + 32 * 864;
    _Float16* pwd1a = pwd0b + 32 * 288;
    _Float16* pwd1b = pwd1a + 64 * 1152;
    _Float16* pwd2a = pwd1b + 64 * 576;
    _Float16* pwd2b = pwd2a + 96 * 1728;
    _Float16* pwu0  = pwd2b + 96 * 864;
    _Float16* pwu1  = pwu0  + 64 * 864;
    _Float16* pwu2  = pwu1  + 32 * 1152;
    _Float16* pwfin = pwu2  + 96 * 576;

    auto g1 = [](size_t n, int b) { return dim3((unsigned)((n + b - 1) / b)); };
    auto pack = [&](const float* w, _Float16* dst, int Cout, int Cin, int Cinp, int Mpad) {
        k_pack_w<<<g1((size_t)Mpad * 9 * Cinp, 256), 256, 0, stream>>>(w, dst, Cout, Cin, Cinp, Mpad);
    };
    auto zb = [&](_Float16* buf, int C, int H, int W) {
        k_zero_border<<<g1((size_t)N * C * (2 * (W + 2) + 2 * H), 256), 256, 0, stream>>>(buf, N, C, H, W);
    };
    auto conv = [&](const _Float16* in, const _Float16* wpk, const float* bias,
                    _Float16* o16, float* o32, int Cinp, int Cout, int Cstore,
                    int Hin, int stride, int relu, int Mpad) {
        int Hout = (stride == 2) ? Hin / 2 : Hin;
        dim3 grid((unsigned)(((size_t)N * Hout * Hout) / CONV_BLKN), (unsigned)(Mpad / MTILE));
        dim3 blk(CONV_THREADS);
#define LC(HH, SS) conv3x3_wmma<HH, HH, SS><<<grid, blk, 0, stream>>>( \
        in, wpk, bias, o16, o32, Cinp, Cout, Cstore, relu)
        if      (Hin == 384 && stride == 2) LC(384, 2);
        else if (Hin == 384 && stride == 1) LC(384, 1);
        else if (Hin == 192 && stride == 2) LC(192, 2);
        else if (Hin == 192 && stride == 1) LC(192, 1);
        else if (Hin == 96  && stride == 2) LC(96, 2);
        else if (Hin == 96  && stride == 1) LC(96, 1);
        else                                LC(48, 1);
#undef LC
    };

    // ---- pack weights (zero-filled to padded shapes) ----
    pack(wd0a, pwd0a, 32, 70, 96, 32);    pack(wd0b, pwd0b, 32, 32, 32, 32);
    pack(wd1a, pwd1a, 64, 128, 128, 64);  pack(wd1b, pwd1b, 64, 64, 64, 64);
    pack(wd2a, pwd2a, 96, 192, 192, 96);  pack(wd2b, pwd2b, 96, 96, 96, 96);
    pack(wu0,  pwu0,  64, 96, 96, 64);    pack(wu1,  pwu1,  32, 128, 128, 32);
    pack(wu2,  pwu2,  70, 64, 64, 96);    pack(wfin, pwfin, 4, 70, 96, 32);

    // ---- encoder ----
    zb(bufA, 96, H1, W1);
    k_cvt_pad<<<g1((size_t)N * 3 * H1 * W1, 256), 256, 0, stream>>>(img0, bufA, N, 3, H1, W1, 0, 96);
    k_cvt_pad<<<g1((size_t)N * 3 * H1 * W1, 256), 256, 0, stream>>>(img1, bufA, N, 3, H1, W1, 3, 96);
    k_warp16<<<g1((size_t)N * H1 * W1, 256), 256, 0, stream>>>(c0_0, ft0, bufA, N, 32, H1, W1, 6, 96);
    k_warp16<<<g1((size_t)N * H1 * W1, 256), 256, 0, stream>>>(c1_0, ft1, bufA, N, 32, H1, W1, 38, 96);
    zb(bufD, 32, H2, W2);
    conv(bufA, pwd0a, bd0a, bufD, nullptr, 96, 32, 32, H1, 2, 1, 32);   // t0 @192
    zb(s0b, 32, H2, W2);
    conv(bufD, pwd0b, bd0b, s0b, nullptr, 32, 32, 32, H2, 1, 1, 32);    // s0

    zb(bufB, 128, H2, W2);
    k_copy16<<<g1((size_t)N * 32 * H2 * W2, 256), 256, 0, stream>>>(s0b, bufB, N, 32, H2, W2, 32, 0, 128);
    k_warp16<<<g1((size_t)N * H2 * W2, 256), 256, 0, stream>>>(c0_1, ft0_s2, bufB, N, 48, H2, W2, 32, 128);
    k_warp16<<<g1((size_t)N * H2 * W2, 256), 256, 0, stream>>>(c1_1, ft1_s2, bufB, N, 48, H2, W2, 80, 128);
    zb(bufE, 64, H4, W4);
    conv(bufB, pwd1a, bd1a, bufE, nullptr, 128, 64, 64, H2, 2, 1, 64);  // t1 @96
    zb(s1b, 64, H4, W4);
    conv(bufE, pwd1b, bd1b, s1b, nullptr, 64, 64, 64, H4, 1, 1, 64);    // s1

    zb(bufD, 192, H4, W4);
    k_copy16<<<g1((size_t)N * 64 * H4 * W4, 256), 256, 0, stream>>>(s1b, bufD, N, 64, H4, W4, 64, 0, 192);
    k_warp16<<<g1((size_t)N * H4 * W4, 256), 256, 0, stream>>>(c0_2, ft0_s4, bufD, N, 64, H4, W4, 64, 192);
    k_warp16<<<g1((size_t)N * H4 * W4, 256), 256, 0, stream>>>(c1_2, ft1_s4, bufD, N, 64, H4, W4, 128, 192);
    zb(bufF, 96, H8, W8);
    conv(bufD, pwd2a, bd2a, bufF, nullptr, 192, 96, 96, H4, 2, 1, 96);  // t2 @48
    zb(s2b, 96, H8, W8);
    conv(bufF, pwd2b, bd2b, s2b, nullptr, 96, 96, 96, H8, 1, 1, 96);    // s2

    // ---- decoder (inner ReLUs are identities: inputs already >= 0) ----
    zb(bufE, 96, H4, W4);
    k_up2x16<<<g1((size_t)N * 96 * H4 * W4, 256), 256, 0, stream>>>(s2b, bufE, N, 96, H8, W8);  // u0_in @96
    zb(bufF, 64, H4, W4);
    conv(bufE, pwu0, bu0, bufF, nullptr, 96, 64, 64, H4, 1, 1, 64);                              // x0 @96

    zb(bufE, 128, H4, W4);
    k_copy16<<<g1((size_t)N * 64 * H4 * W4, 256), 256, 0, stream>>>(bufF, bufE, N, 64, H4, W4, 64, 0, 128);
    k_copy16<<<g1((size_t)N * 64 * H4 * W4, 256), 256, 0, stream>>>(s1b,  bufE, N, 64, H4, W4, 64, 64, 128);
    zb(bufD, 128, H2, W2);
    k_up2x16<<<g1((size_t)N * 128 * H2 * W2, 256), 256, 0, stream>>>(bufE, bufD, N, 128, H4, W4); // up1 @192
    zb(bufF, 32, H2, W2);
    conv(bufD, pwu1, bu1, bufF, nullptr, 128, 32, 32, H2, 1, 1, 32);                              // x1 @192

    zb(bufB, 64, H2, W2);
    k_copy16<<<g1((size_t)N * 32 * H2 * W2, 256), 256, 0, stream>>>(bufF, bufB, N, 32, H2, W2, 32, 0, 64);
    k_copy16<<<g1((size_t)N * 32 * H2 * W2, 256), 256, 0, stream>>>(s0b,  bufB, N, 32, H2, W2, 32, 32, 64);
    zb(bufA, 64, H1, W1);
    k_up2x16<<<g1((size_t)N * 64 * H1 * W1, 256), 256, 0, stream>>>(bufB, bufA, N, 64, H2, W2);   // up2 @384
    zb(bufB, 96, H1, W1);
    conv(bufA, pwu2, bu2, bufB, nullptr, 64, 70, 96, H1, 1, 1, 96);                               // x2 @384

    conv(bufB, pwfin, bfin, nullptr, (float*)d_out, 96, 4, 4, H1, 1, 0, 32);                      // [N,4,384,384]
}
